// PerceiverBlock_modified_53403623359030
// MI455X (gfx1250) — matrix operand order repaired and checked by reading, not compiled, not run on hardware
//
#include <hip/hip_runtime.h>
#include <hip/hip_bf16.h>
#include <math.h>

typedef __attribute__((ext_vector_type(16))) _Float16 v16h;
typedef __attribute__((ext_vector_type(8)))  _Float16 v8h;
typedef __attribute__((ext_vector_type(8)))  float    v8f;
typedef __attribute__((ext_vector_type(4)))  unsigned int u32x4;
typedef __attribute__((ext_vector_type(8)))  unsigned int u32x8;
typedef int v4i_vs __attribute__((vector_size(16)));   // matches builtin param type

#define DIM   1024
#define HEADS 16
#define DH    64
#define INNER 1024
#define FF    4096
#define BATCH 4
#define N1    1024
#define N2    256
#define NTOK  1280          // N1 + N2
#define MTOK  (BATCH*NTOK)  // 5120
#define SCALE 0.125f        // 64^-0.5
#define LNEPS 1e-5f

#if __has_builtin(__builtin_amdgcn_global_load_async_to_lds_b128)
#define HAVE_ASYNC_LDS 1
#endif

// ---------------------------------------------------------------- helpers
__device__ __forceinline__ v16h join16(v8h lo, v8h hi) {
  v16h r;
#pragma unroll
  for (int i = 0; i < 8; ++i) { r[i] = lo[i]; r[i + 8] = hi[i]; }
  return r;
}

// 16-byte global -> LDS copy; async DMA when the toolchain exposes it.
__device__ __forceinline__ void async_copy16(const _Float16* g, _Float16* l) {
#ifdef HAVE_ASYNC_LDS
  __builtin_amdgcn_global_load_async_to_lds_b128(
      (__attribute__((address_space(1))) v4i_vs*)g,
      (__attribute__((address_space(3))) v4i_vs*)l, 0, 0);
#else
  *(v8h*)l = *(const v8h*)g;
#endif
}

__device__ __forceinline__ void wait_async() {
#ifdef HAVE_ASYNC_LDS
#if __has_builtin(__builtin_amdgcn_s_wait_asynccnt)
  __builtin_amdgcn_s_wait_asynccnt(0);
#else
  asm volatile("s_wait_asynccnt 0" ::: "memory");
#endif
#endif
}

// LDS byte offset of a generic shared pointer (LDS aperture: low 32 bits of the address)
__device__ __forceinline__ unsigned lds_off_of(const void* p) {
  return (unsigned)(unsigned long long)(size_t)p;
}

// Tensor Data Mover: load a 64-row x 256-col f16 tile (row stride Nn elements)
// from global into LDS (contiguous row-major tile). Issued once per wave.
__device__ __forceinline__ void tdm_load_b_tile(const _Float16* gsrc, unsigned lds_addr,
                                                int Nn, int K) {
  unsigned long long ga = (unsigned long long)(size_t)gsrc;
  u32x4 g0;
  g0[0] = 1u;                                         // count=1 user descriptor
  g0[1] = lds_addr;                                   // lds_addr [63:32]
  g0[2] = (unsigned)(ga & 0xffffffffu);               // global_addr lo
  g0[3] = (unsigned)((ga >> 32) & 0x01ffffffu) | (2u << 30);  // addr hi + type=2
  u32x8 g1;
  g1[0] = (1u << 16);                                 // data_size = 2 bytes
  g1[1] = ((unsigned)Nn & 0xffffu) << 16;             // tensor_dim0 lo16
  g1[2] = ((unsigned)Nn >> 16) | (((unsigned)K & 0xffffu) << 16);  // dim0 hi, dim1 lo
  g1[3] = ((unsigned)K >> 16) | (256u << 16);         // dim1 hi, tile_dim0=256
  g1[4] = 64u;                                        // tile_dim1=64, tile_dim2=0
  g1[5] = (unsigned)Nn;                               // tensor_dim0_stride lo32
  g1[6] = 0u;
  g1[7] = 0u;
  asm volatile("tensor_load_to_lds %0, %1" :: "s"(g0), "s"(g1) : "memory");
}

__device__ __forceinline__ void wait_tensor() {
#if __has_builtin(__builtin_amdgcn_s_wait_tensorcnt)
  __builtin_amdgcn_s_wait_tensorcnt(0);
#else
  asm volatile("s_wait_tensorcnt 0" ::: "memory");
#endif
}

// ---------------------------------------------------------------- weights f32 -> f16
__global__ __launch_bounds__(256) void k_f32_to_f16(const float* __restrict__ src,
                                                    _Float16* __restrict__ dst, int n) {
  int i = blockIdx.x * 256 + threadIdx.x;
  if (i < n) dst[i] = (_Float16)src[i];
}

// ---------------------------------------------------------------- LN(x)/LN(latents) -> f16 qkv_in
__global__ __launch_bounds__(256) void k_ln_qkvin(const float* __restrict__ x,
                                                  const float* __restrict__ lat,
                                                  const float* __restrict__ gm, const float* __restrict__ bm,
                                                  const float* __restrict__ gl, const float* __restrict__ bl,
                                                  _Float16* __restrict__ out16) {
  __shared__ float red[256];
  int r  = blockIdx.x;            // 0..MTOK-1
  int bi = r / NTOK, lr = r % NTOK;
  const float *src, *g, *bb;
  if (lr < N1) { src = x   + ((size_t)bi * N1 + lr)        * DIM; g = gm; bb = bm; }
  else         { src = lat + ((size_t)bi * N2 + (lr - N1)) * DIM; g = gl; bb = bl; }
  int t = threadIdx.x;
  float v[4], s = 0.f, sq = 0.f;
#pragma unroll
  for (int i = 0; i < 4; ++i) { v[i] = src[t + i * 256]; s += v[i]; sq += v[i] * v[i]; }
  red[t] = s; __syncthreads();
  for (int st = 128; st > 0; st >>= 1) { if (t < st) red[t] += red[t + st]; __syncthreads(); }
  float mean = red[0] * (1.f / DIM); __syncthreads();
  red[t] = sq; __syncthreads();
  for (int st = 128; st > 0; st >>= 1) { if (t < st) red[t] += red[t + st]; __syncthreads(); }
  float var  = red[0] * (1.f / DIM) - mean * mean;
  float rstd = rsqrtf(var + LNEPS);
  _Float16* dst = out16 + (size_t)r * DIM;
#pragma unroll
  for (int i = 0; i < 4; ++i) {
    int c = t + i * 256;
    dst[c] = (_Float16)((v[i] - mean) * rstd * g[c] + bb[c]);
  }
}

// ---------------------------------------------------------------- tiled WMMA GEMM
// C[M,Nn] = A[M,K] @ B[K,Nn]; A,B f16 row-major. mode: 0=f32 out, 1=f16 out, 2=gelu->f16
// Block tile 128x256, BK=64; 8 waves, each 64x64 (16 WMMA accumulators).
__global__ __launch_bounds__(256) void k_gemm(const _Float16* __restrict__ A,
                                              const _Float16* __restrict__ Bw,
                                              float* __restrict__ Cf,
                                              _Float16* __restrict__ Ch,
                                              int M, int Nn, int K, int mode) {
  __shared__ _Float16 As[128 * 64];   // 16 KB
  __shared__ _Float16 Bs[64 * 256];   // 32 KB
  int t = threadIdx.x, w = t >> 5, lane = t & 31;
  int wm = (w >> 2) * 64, wn = (w & 3) * 64;
  int bm = blockIdx.y * 128, bn = blockIdx.x * 256;
  v8f acc[4][4] = {};
  int mrow = lane & 15;
  int koff = (lane >> 4) * 8;
  int arow = t >> 1, aseg = (t & 1) * 32;    // A: 128 rows x 64 cols, 32 halves/thread
  unsigned bs_lds = lds_off_of(Bs);

  for (int k0 = 0; k0 < K; k0 += 64) {
    { // stage A via per-lane async b128 DMA
      const _Float16* ga = A + (size_t)(bm + arow) * K + k0 + aseg;
      _Float16* la = As + arow * 64 + aseg;
#pragma unroll
      for (int i = 0; i < 4; ++i) async_copy16(ga + i * 8, la + i * 8);
      // prefetch next K-slice of A
      int kn = (k0 + 64 < K) ? (k0 + 64) : k0;
      __builtin_prefetch(A + (size_t)(bm + arow) * K + kn + aseg, 0, 1);
    }
    // stage B via Tensor Data Mover (one DMA for the whole 64x256 tile)
    if (w == 0) {
      tdm_load_b_tile(Bw + (size_t)k0 * Nn + bn, bs_lds, Nn, K);
      wait_tensor();
    }
    wait_async();
    __syncthreads();

#pragma unroll
    for (int c = 0; c < 2; ++c) {
      v16h af[4];
#pragma unroll
      for (int mt = 0; mt < 4; ++mt) {
        const _Float16* p = As + (wm + mt * 16 + mrow) * 64 + c * 32;
        af[mt] = join16(*(const v8h*)(p + koff), *(const v8h*)(p + 16 + koff));
      }
#pragma unroll
      for (int nt = 0; nt < 4; ++nt) {
        const _Float16* p = Bs + (size_t)(c * 32 + koff) * 256 + wn + nt * 16 + mrow;
        v16h bf;
#pragma unroll
        for (int i = 0; i < 8; ++i) { bf[i] = p[i * 256]; bf[i + 8] = p[(16 + i) * 256]; }
#pragma unroll
        for (int mt = 0; mt < 4; ++mt)
          acc[mt][nt] = __builtin_amdgcn_wmma_f32_16x16x32_f16(
              false, af[mt], false, bf, (short)0, acc[mt][nt], false, false);
      }
    }
    __syncthreads();
  }

  int mbase = (lane >> 4) * 8, col_l = lane & 15;
#pragma unroll
  for (int mt = 0; mt < 4; ++mt)
#pragma unroll
    for (int nt = 0; nt < 4; ++nt) {
      int colg = bn + wn + nt * 16 + col_l;
#pragma unroll
      for (int r = 0; r < 8; ++r) {
        int rowg = bm + wm + mt * 16 + mbase + r;
        float v = acc[mt][nt][r];
        if (mode == 0) {
          Cf[(size_t)rowg * Nn + colg] = v;
        } else {
          if (mode == 2) v = 0.5f * v * (1.f + erff(v * 0.70710678118f));
          Ch[(size_t)rowg * Nn + colg] = (_Float16)v;
        }
      }
    }
}

// ---------------------------------------------------------------- flash attention (per b,h,128-row block)
__global__ __launch_bounds__(256) void k_attn(const _Float16* __restrict__ q16,
                                              const _Float16* __restrict__ kv16,
                                              _Float16* __restrict__ attn16) {
  __shared__ _Float16 Ktt[64 * 32];      // [d][token] (transposed)
  __shared__ _Float16 Vt[32 * 64];       // [token][d]
  __shared__ _Float16 Ps[8 * 16 * 32];   // per-wave P scratch
  int t = threadIdx.x, w = t >> 5, lane = t & 31;
  int blk = blockIdx.x;
  int mblk = blk % 10;
  int h    = (blk / 10) % HEADS;
  int bi   = blk / (10 * HEADS);
  int qrow0 = mblk * 128 + w * 16;
  const size_t KVLD = 2 * INNER;
  int mrow = lane & 15, koff = (lane >> 4) * 8;

  v16h aq[2];
#pragma unroll
  for (int c = 0; c < 2; ++c) {
    const _Float16* p = q16 + ((size_t)(bi * NTOK) + qrow0 + mrow) * INNER + h * DH + c * 32;
    aq[c] = join16(*(const v8h*)(p + koff), *(const v8h*)(p + 16 + koff));
  }

  v8f o[4] = {};
  float mrun[8], lrun[8];
#pragma unroll
  for (int r = 0; r < 8; ++r) { mrun[r] = -1e30f; lrun[r] = 0.f; }

  for (int jt = 0; jt < 40; ++jt) {
    int j0 = jt * 32;
    { // cooperative K (transposed, manual) + V (async DMA) stage
      int tok = t >> 3, d0 = (t & 7) * 8;
      const _Float16* vp = kv16 + ((size_t)(bi * NTOK) + j0 + tok) * KVLD + INNER + h * DH + d0;
      async_copy16(vp, Vt + tok * 64 + d0);
      const _Float16* kp = kv16 + ((size_t)(bi * NTOK) + j0 + tok) * KVLD + h * DH + d0;
      v8h kx = *(const v8h*)kp;
#pragma unroll
      for (int i = 0; i < 8; ++i) Ktt[(d0 + i) * 32 + tok] = kx[i];
    }
    wait_async();
    __syncthreads();

    // S = Q K^T (two 16x16 tiles over 32 tokens, K-dim 64 = 2 chunks)
    v8f sacc[2] = {};
#pragma unroll
    for (int sub = 0; sub < 2; ++sub)
#pragma unroll
      for (int c = 0; c < 2; ++c) {
        const _Float16* p = Ktt + (size_t)(c * 32 + koff) * 32 + sub * 16 + mrow;
        v16h bk;
#pragma unroll
        for (int i = 0; i < 8; ++i) { bk[i] = p[i * 32]; bk[i + 8] = p[(16 + i) * 32]; }
        sacc[sub] = __builtin_amdgcn_wmma_f32_16x16x32_f16(
            false, aq[c], false, bk, (short)0, sacc[sub], false, false);
      }

    // online softmax over 32 columns (row = mbase+r within 16-lane group)
#pragma unroll
    for (int r = 0; r < 8; ++r) {
      float a = sacc[0][r] * SCALE, b = sacc[1][r] * SCALE;
      sacc[0][r] = a; sacc[1][r] = b;
      float mx = fmaxf(a, b);
#pragma unroll
      for (int d = 1; d < 16; d <<= 1) mx = fmaxf(mx, __shfl_xor(mx, d, 32));
      float mnew = fmaxf(mrun[r], mx);
      float p0 = __expf(a - mnew), p1 = __expf(b - mnew);
      sacc[0][r] = p0; sacc[1][r] = p1;
      float rs = p0 + p1;
#pragma unroll
      for (int d = 1; d < 16; d <<= 1) rs += __shfl_xor(rs, d, 32);
      float alpha = __expf(mrun[r] - mnew);
      lrun[r] = lrun[r] * alpha + rs;
      mrun[r] = mnew;
#pragma unroll
      for (int nt = 0; nt < 4; ++nt) o[nt][r] *= alpha;
    }

    // reshape P (C-layout) -> A-layout via per-wave LDS
    {
      _Float16* pp = Ps + w * 16 * 32;
      int mbase = (lane >> 4) * 8, nn = lane & 15;
#pragma unroll
      for (int r = 0; r < 8; ++r) {
        pp[(mbase + r) * 32 + nn]      = (_Float16)sacc[0][r];
        pp[(mbase + r) * 32 + 16 + nn] = (_Float16)sacc[1][r];
      }
    }
    v16h pa;
    {
      const _Float16* pp = Ps + w * 16 * 32 + mrow * 32;
      pa = join16(*(const v8h*)(pp + koff), *(const v8h*)(pp + 16 + koff));
    }
    // O += P @ V
#pragma unroll
    for (int nt = 0; nt < 4; ++nt) {
      const _Float16* vp = Vt + koff * 64 + nt * 16 + mrow;
      v16h bv;
#pragma unroll
      for (int i = 0; i < 8; ++i) { bv[i] = vp[i * 64]; bv[i + 8] = vp[(16 + i) * 64]; }
      o[nt] = __builtin_amdgcn_wmma_f32_16x16x32_f16(
          false, pa, false, bv, (short)0, o[nt], false, false);
    }
    __syncthreads();
  }

  { // finalize: O / l -> f16
    int mbase = (lane >> 4) * 8, nn = lane & 15;
#pragma unroll
    for (int nt = 0; nt < 4; ++nt)
#pragma unroll
      for (int r = 0; r < 8; ++r) {
        float v = o[nt][r] / lrun[r];
        attn16[((size_t)(bi * NTOK) + qrow0 + mbase + r) * INNER + h * DH + nt * 16 + nn] = (_Float16)v;
      }
  }
}

// ---------------------------------------------------------------- proj + residual -> resid_out(f32), LN -> f16
__global__ __launch_bounds__(256) void k_resid_ln(const float* __restrict__ proj,
                                                  const float* __restrict__ x,
                                                  const float* __restrict__ lat,
                                                  const float* __restrict__ gf, const float* __restrict__ bf,
                                                  float* __restrict__ resid_out,
                                                  _Float16* __restrict__ ln16) {
  __shared__ float red[256];
  int r  = blockIdx.x;
  int bi = r / NTOK, lr = r % NTOK;
  const float* res = (lr < N1) ? x   + ((size_t)bi * N1 + lr)        * DIM
                               : lat + ((size_t)bi * N2 + (lr - N1)) * DIM;
  const float* pr  = proj + (size_t)r * DIM;
  float*       ro  = resid_out + (size_t)r * DIM;
  int t = threadIdx.x;
  float v[4], s = 0.f, sq = 0.f;
#pragma unroll
  for (int i = 0; i < 4; ++i) {
    int c = t + i * 256;
    v[i] = pr[c] + res[c];
    ro[c] = v[i];
    s += v[i]; sq += v[i] * v[i];
  }
  red[t] = s; __syncthreads();
  for (int st = 128; st > 0; st >>= 1) { if (t < st) red[t] += red[t + st]; __syncthreads(); }
  float mean = red[0] * (1.f / DIM); __syncthreads();
  red[t] = sq; __syncthreads();
  for (int st = 128; st > 0; st >>= 1) { if (t < st) red[t] += red[t + st]; __syncthreads(); }
  float var  = red[0] * (1.f / DIM) - mean * mean;
  float rstd = rsqrtf(var + LNEPS);
  _Float16* dst = ln16 + (size_t)r * DIM;
#pragma unroll
  for (int i = 0; i < 4; ++i) {
    int c = t + i * 256;
    dst[c] = (_Float16)((v[i] - mean) * rstd * gf[c] + bf[c]);
  }
}

// ---------------------------------------------------------------- final: ff2 + resid -> d_out (split order)
__global__ __launch_bounds__(256) void k_final(const float* __restrict__ ff2,
                                               const float* __restrict__ resid,
                                               float* __restrict__ out) {
  size_t idx = (size_t)blockIdx.x * 256 + threadIdx.x;
  int r = (int)(idx / DIM), c = (int)(idx % DIM);
  int bi = r / NTOK, lr = r % NTOK;
  size_t off = (lr < N1) ? ((size_t)(bi * N1 + lr) * DIM + c)
                         : ((size_t)BATCH * N1 * DIM + (size_t)(bi * N2 + (lr - N1)) * DIM + c);
  out[off] = ff2[idx] + resid[idx];
}

// ---------------------------------------------------------------- host driver
extern "C" void kernel_launch(void* const* d_in, const int* in_sizes, int n_in,
                              void* d_out, int out_size, void* d_ws, size_t ws_size,
                              hipStream_t stream) {
  (void)in_sizes; (void)n_in; (void)out_size; (void)ws_size;
  const float* x   = (const float*)d_in[0];
  const float* lat = (const float*)d_in[1];
  const float* Wq  = (const float*)d_in[2];
  const float* Wkv = (const float*)d_in[3];
  const float* Wo  = (const float*)d_in[4];
  const float* W1  = (const float*)d_in[5];
  const float* W2  = (const float*)d_in[6];
  const float* gm  = (const float*)d_in[7];
  const float* bm  = (const float*)d_in[8];
  const float* gl  = (const float*)d_in[9];
  const float* bl  = (const float*)d_in[10];
  const float* gf  = (const float*)d_in[11];
  const float* bf  = (const float*)d_in[12];
  float* out = (float*)d_out;

  // workspace carve-up (256B aligned)
  char* ws = (char*)d_ws; size_t off = 0;
  auto carve = [&](size_t bytes) { void* p = ws + off; off += (bytes + 255) & ~(size_t)255; return p; };
  _Float16* qkv16  = (_Float16*)carve((size_t)MTOK * DIM * 2);
  _Float16* Wq16   = (_Float16*)carve((size_t)DIM * INNER * 2);
  _Float16* Wkv16  = (_Float16*)carve((size_t)DIM * 2 * INNER * 2);
  _Float16* Wo16   = (_Float16*)carve((size_t)INNER * DIM * 2);
  _Float16* W116   = (_Float16*)carve((size_t)DIM * FF * 2);
  _Float16* W216   = (_Float16*)carve((size_t)FF * DIM * 2);
  _Float16* q16    = (_Float16*)carve((size_t)MTOK * INNER * 2);
  _Float16* kv16   = (_Float16*)carve((size_t)MTOK * 2 * INNER * 2);
  _Float16* attn16 = (_Float16*)carve((size_t)MTOK * INNER * 2);
  float*    proj   = (float*)   carve((size_t)MTOK * DIM * 4);
  float*    resid  = (float*)   carve((size_t)MTOK * DIM * 4);
  _Float16* ln16   = (_Float16*)carve((size_t)MTOK * DIM * 2);
  _Float16* ff116  = (_Float16*)carve((size_t)MTOK * FF * 2);
  float*    ff2    = (float*)   carve((size_t)MTOK * DIM * 4);

  // 1. convert weights to f16
  k_f32_to_f16<<<(DIM * INNER) / 256, 256, 0, stream>>>(Wq, Wq16, DIM * INNER);
  k_f32_to_f16<<<(DIM * 2 * INNER) / 256, 256, 0, stream>>>(Wkv, Wkv16, DIM * 2 * INNER);
  k_f32_to_f16<<<(INNER * DIM) / 256, 256, 0, stream>>>(Wo, Wo16, INNER * DIM);
  k_f32_to_f16<<<(DIM * FF) / 256, 256, 0, stream>>>(W1, W116, DIM * FF);
  k_f32_to_f16<<<(FF * DIM) / 256, 256, 0, stream>>>(W2, W216, FF * DIM);

  // 2. layernorm x / latents -> qkv_in (f16)
  k_ln_qkvin<<<MTOK, 256, 0, stream>>>(x, lat, gm, bm, gl, bl, qkv16);

  // 3. Q = qkv_in @ Wq (f16 out); KV = qkv_in @ Wkv (f16 out)
  k_gemm<<<dim3(INNER / 256, MTOK / 128), 256, 0, stream>>>(qkv16, Wq16, nullptr, q16, MTOK, INNER, DIM, 1);
  k_gemm<<<dim3((2 * INNER) / 256, MTOK / 128), 256, 0, stream>>>(qkv16, Wkv16, nullptr, kv16, MTOK, 2 * INNER, DIM, 1);

  // 4. flash attention: (b, h, 10 row-blocks)
  k_attn<<<BATCH * HEADS * 10, 256, 0, stream>>>(q16, kv16, attn16);

  // 5. proj = attn @ Wo (f32 out)
  k_gemm<<<dim3(DIM / 256, MTOK / 128), 256, 0, stream>>>(attn16, Wo16, proj, nullptr, MTOK, DIM, INNER, 0);

  // 6. resid_out = proj + residual; ln16 = LN(resid_out)
  k_resid_ln<<<MTOK, 256, 0, stream>>>(proj, x, lat, gf, bf, resid, ln16);

  // 7. ff1 = gelu(ln16 @ W1) (f16 out); ff2 = ff1 @ W2 (f32 out)
  k_gemm<<<dim3(FF / 256, MTOK / 128), 256, 0, stream>>>(ln16, W116, nullptr, ff116, MTOK, FF, DIM, 2);
  k_gemm<<<dim3(DIM / 256, MTOK / 128), 256, 0, stream>>>(ff116, W216, ff2, nullptr, MTOK, DIM, FF, 0);

  // 8. out = ff2 + resid_out, in (x-part, latent-part) flat order
  k_final<<<((size_t)MTOK * DIM) / 256, 256, 0, stream>>>(ff2, resid, out);
}
